// MultiHeadQKVAttention_41936060678242
// MI455X (gfx1250) — matrix operand order repaired and checked by reading, not compiled, not run on hardware
//
#include <hip/hip_runtime.h>

typedef __attribute__((ext_vector_type(16))) _Float16 v16h;
typedef __attribute__((ext_vector_type(8)))  _Float16 v8h;
typedef __attribute__((ext_vector_type(8)))  float    v8f;

#define B_    8
#define M_    2048
#define N_    2048
#define DK_   256
#define DV_   32      // per-head value dim
#define DOUT_ 256
#define H_    8

__device__ __forceinline__ float fast_exp2(float x) {
#if defined(__has_builtin)
#if __has_builtin(__builtin_amdgcn_exp2f)
    return __builtin_amdgcn_exp2f(x);
#else
    return exp2f(x);
#endif
#else
    return exp2f(x);
#endif
}

// ---- Prologue: keys f32 -> f16 (row-major), 8 elements/thread ----
__global__ void convert_k_kernel(const float* __restrict__ k, _Float16* __restrict__ kf) {
    const size_t i = ((size_t)blockIdx.x * blockDim.x + threadIdx.x) * 8;
    const float4* s = (const float4*)(k + i);
    float4 a = s[0], b = s[1];
    v8h o;
    o[0] = (_Float16)a.x; o[1] = (_Float16)a.y; o[2] = (_Float16)a.z; o[3] = (_Float16)a.w;
    o[4] = (_Float16)b.x; o[5] = (_Float16)b.y; o[6] = (_Float16)b.z; o[7] = (_Float16)b.w;
    *(v8h*)(kf + i) = o;
}

// ---- Prologue: values f32 [b][n][vf] -> f16 transposed [b][vf][n] ----
__global__ void convert_v_kernel(const float* __restrict__ v, _Float16* __restrict__ vt) {
    const int idx = blockIdx.x * 256 + threadIdx.x;   // 0 .. B*DV*N-1
    const int n  = idx & (N_ - 1);
    const int t  = idx >> 11;
    const int vf = t & (DV_ - 1);
    const int b  = t >> 5;
    vt[((size_t)b * DV_ + vf) * N_ + n] = (_Float16)v[((size_t)b * N_ + n) * DV_ + vf];
}

// One wave (32 lanes) owns a 16-row tile of M; flash attention over N in 64-col tiles.
// 256 threads/block = 8 waves; grid = B*M/16/8 = 128 blocks.
template <bool PRE>
__launch_bounds__(256)
__global__ void mha_flash_kernel(const float* __restrict__ q,
                                 const float* __restrict__ kk,
                                 const float* __restrict__ vv,
                                 const float* __restrict__ qp,
                                 const float* __restrict__ kpr,
                                 const float* __restrict__ wo,
                                 const float* __restrict__ bo,
                                 const _Float16* __restrict__ kf16,   // [b][n][k] f16
                                 const _Float16* __restrict__ vf16t,  // [b][vf][n] f16
                                 float* __restrict__ out)
{
    __shared__ _Float16 weff[DOUT_ * DV_];   // folded output weights, 16 KB
    __shared__ _Float16 pbuf[8][16 * 64];    // per-wave P / head scratch, 16 KB

    const int tid = threadIdx.x;

    // ---- Fold w_o[256,256] over the 8 identical-head tiles: weff[d,v] = sum_h wo[d, 32h+v]
    {
        const int d = tid;  // 0..255
        for (int v = 0; v < DV_; ++v) {
            float s = 0.f;
            #pragma unroll
            for (int h = 0; h < H_; ++h) s += wo[d * (DV_ * H_) + h * DV_ + v];
            weff[d * DV_ + v] = (_Float16)s;
        }
    }
    __syncthreads();

    const int wv   = tid >> 5;
    const int lane = tid & 31;
    const int nn   = lane & 15;   // row index (A-layout) / column index (B,C layout)
    const int hi   = lane >> 4;

    const int wid = blockIdx.x * 8 + wv;     // 0..1023
    const int b   = wid >> 7;                // batch
    const int m0  = (wid & 127) << 4;        // row-tile base

    // ---- Load Q tile (16x256) into A-layout regs ----
    // pre-scale by q_pres * (1/sqrt(32)) * log2(e): softmax done in exp2 domain
    const float qs = qp[b * M_ + m0 + nn] * 0.17677669529663687f * 1.4426950408889634f;
    const float* qrow = q + (size_t)(b * M_ + m0 + nn) * DK_;
    v16h qa[8];
    #pragma unroll
    for (int c = 0; c < 8; ++c) {
        // A elem e -> K = 32c + 16*hi + e : one contiguous run of 16
        const float4* p0 = (const float4*)(qrow + 32 * c + 16 * hi);
        const float4* p1 = (const float4*)(qrow + 32 * c + 16 * hi + 8);
        float4 a0 = p0[0], a1 = p0[1];
        float4 b0 = p1[0], b1 = p1[1];
        qa[c][0]  = (_Float16)(a0.x * qs); qa[c][1]  = (_Float16)(a0.y * qs);
        qa[c][2]  = (_Float16)(a0.z * qs); qa[c][3]  = (_Float16)(a0.w * qs);
        qa[c][4]  = (_Float16)(a1.x * qs); qa[c][5]  = (_Float16)(a1.y * qs);
        qa[c][6]  = (_Float16)(a1.z * qs); qa[c][7]  = (_Float16)(a1.w * qs);
        qa[c][8]  = (_Float16)(b0.x * qs); qa[c][9]  = (_Float16)(b0.y * qs);
        qa[c][10] = (_Float16)(b0.z * qs); qa[c][11] = (_Float16)(b0.w * qs);
        qa[c][12] = (_Float16)(b1.x * qs); qa[c][13] = (_Float16)(b1.y * qs);
        qa[c][14] = (_Float16)(b1.z * qs); qa[c][15] = (_Float16)(b1.w * qs);
    }

    float rmax[8], lsum[8];
    #pragma unroll
    for (int g = 0; g < 8; ++g) { rmax[g] = -1e30f; lsum[g] = 0.f; }
    v8f o0 = {}; v8f o1 = {};

    const float*    kbase  = kk + (size_t)b * N_ * DK_;
    const float*    vbase  = vv + (size_t)b * N_ * DV_;
    const _Float16* kfbase = PRE ? (kf16 + (size_t)b * N_ * DK_) : nullptr;
    const _Float16* vtbase = PRE ? (vf16t + ((size_t)b * DV_ + nn) * N_) : nullptr;
    const float* kpb = kpr + b * N_;
    _Float16* myp = &pbuf[wv][0];

    for (int n0 = 0; n0 < N_; n0 += 64) {
        // ---- S = Q K^T for 64 columns (four 16-wide halves), K-dim = 256 ----
        v8f sAcc[4];
        #pragma unroll
        for (int s = 0; s < 4; ++s) {
            v8f acc = {};
            if constexpr (PRE) {
                const _Float16* krow = kfbase + (size_t)(n0 + 16 * s + nn) * DK_ + 16 * hi;
                #pragma unroll
                for (int c = 0; c < 8; ++c) {
                    const v16h kb = *(const v16h*)(krow + 32 * c);  // 32 B contiguous
                    acc = __builtin_amdgcn_wmma_f32_16x16x32_f16(
                        false, qa[c], false, kb, (short)0, acc, false, false);
                }
            } else {
                const float* krow = kbase + (size_t)(n0 + 16 * s + nn) * DK_ + 16 * hi;
                #pragma unroll
                for (int c = 0; c < 8; ++c) {
                    const float4* kp4 = (const float4*)(krow + 32 * c);
                    v16h kb;
                    #pragma unroll
                    for (int e4 = 0; e4 < 4; ++e4) {
                        float4 x = kp4[e4];
                        kb[4 * e4 + 0] = (_Float16)x.x;
                        kb[4 * e4 + 1] = (_Float16)x.y;
                        kb[4 * e4 + 2] = (_Float16)x.z;
                        kb[4 * e4 + 3] = (_Float16)x.w;
                    }
                    acc = __builtin_amdgcn_wmma_f32_16x16x32_f16(
                        false, qa[c], false, kb, (short)0, acc, false, false);
                }
            }
            sAcc[s] = acc;
        }

        float km[4];
        #pragma unroll
        for (int s = 0; s < 4; ++s) km[s] = kpb[n0 + 16 * s + nn];

        // masked logits (k_pres==0 -> -inf)
        float pv[4][8];
        #pragma unroll
        for (int s = 0; s < 4; ++s) {
            #pragma unroll
            for (int g = 0; g < 8; ++g)
                pv[s][g] = (km[s] > 0.5f) ? sAcc[s][g] : -1e30f;
        }

        // ---- online softmax: one reduce per row per 64 columns ----
        #pragma unroll
        for (int g = 0; g < 8; ++g) {
            float bm = fmaxf(fmaxf(pv[0][g], pv[1][g]), fmaxf(pv[2][g], pv[3][g]));
            bm = fmaxf(bm, __shfl_xor(bm, 1, 32));
            bm = fmaxf(bm, __shfl_xor(bm, 2, 32));
            bm = fmaxf(bm, __shfl_xor(bm, 4, 32));
            bm = fmaxf(bm, __shfl_xor(bm, 8, 32));
            const float nm    = fmaxf(rmax[g], bm);
            const float alpha = fast_exp2(rmax[g] - nm);
            rmax[g] = nm;
            o0[g] *= alpha;
            o1[g] *= alpha;
            lsum[g] *= alpha;
            float rs = 0.f;
            #pragma unroll
            for (int s = 0; s < 4; ++s) {
                pv[s][g] = fast_exp2(pv[s][g] - nm);
                rs += pv[s][g];
                // C-layout -> row-major LDS: row = g + 8*hi, col = 16*s + nn
                myp[(g + 8 * hi) * 64 + 16 * s + nn] = (_Float16)pv[s][g];
            }
            rs += __shfl_xor(rs, 1, 32);
            rs += __shfl_xor(rs, 2, 32);
            rs += __shfl_xor(rs, 4, 32);
            rs += __shfl_xor(rs, 8, 32);
            lsum[g] += rs;
        }

        // ---- reload P in A-layout: two 16x32 tiles ----
        v16h pa0, pa1;
        {
            const _Float16* pr0 = myp + nn * 64 + 16 * hi;       // cols 0..31
            const _Float16* pr1 = pr0 + 32;                      // cols 32..63
            v8h a0 = *(const v8h*)(pr0), a1 = *(const v8h*)(pr0 + 8);
            v8h b0v = *(const v8h*)(pr1), b1v = *(const v8h*)(pr1 + 8);
            #pragma unroll
            for (int i = 0; i < 8; ++i) {
                pa0[i] = a0[i]; pa0[i + 8] = a1[i];
                pa1[i] = b0v[i]; pa1[i + 8] = b1v[i];
            }
        }

        // ---- V tiles in B-layout: B[k'=16*hi+e, vf=nn(+16)], two 32-row blocks ----
        v16h vb0a, vb0b, vb1a, vb1b;
        if constexpr (PRE) {
            const _Float16* vt0 = vtbase + n0 + 16 * hi;          // vf = nn
            vb0a = *(const v16h*)(vt0);
            vb0b = *(const v16h*)(vt0 + 32);
            vb1a = *(const v16h*)(vt0 + 16 * N_);                 // vf = nn + 16
            vb1b = *(const v16h*)(vt0 + 16 * N_ + 32);
        } else {
            const float* vrow = vbase + (size_t)(n0 + 16 * hi) * DV_;
            #pragma unroll
            for (int e = 0; e < 16; ++e) {
                vb0a[e] = (_Float16)vrow[e * DV_ + nn];
                vb1a[e] = (_Float16)vrow[e * DV_ + 16 + nn];
                vb0b[e] = (_Float16)vrow[(e + 32) * DV_ + nn];
                vb1b[e] = (_Float16)vrow[(e + 32) * DV_ + 16 + nn];
            }
        }
        o0 = __builtin_amdgcn_wmma_f32_16x16x32_f16(false, pa0, false, vb0a, (short)0, o0, false, false);
        o1 = __builtin_amdgcn_wmma_f32_16x16x32_f16(false, pa0, false, vb1a, (short)0, o1, false, false);
        o0 = __builtin_amdgcn_wmma_f32_16x16x32_f16(false, pa1, false, vb0b, (short)0, o0, false, false);
        o1 = __builtin_amdgcn_wmma_f32_16x16x32_f16(false, pa1, false, vb1b, (short)0, o1, false, false);
    }

    // ---- normalize head, stage to LDS as f16 ----
    #pragma unroll
    for (int g = 0; g < 8; ++g) {
        const float inv = 1.0f / lsum[g];
        myp[(g + 8 * hi) * 64 + nn]      = (_Float16)(o0[g] * inv);
        myp[(g + 8 * hi) * 64 + 16 + nn] = (_Float16)(o1[g] * inv);
    }

    // ---- head (A-layout) once; Out(16x256) = head(16x32) @ weff^T + b_o via WMMA ----
    const _Float16* hr = myp + nn * 64 + 16 * hi;
    v8h hlo = *(const v8h*)(hr);
    v8h hhi = *(const v8h*)(hr + 8);
    v16h ha;
    #pragma unroll
    for (int i = 0; i < 8; ++i) { ha[i] = hlo[i]; ha[i + 8] = hhi[i]; }

    float* orow = out + (size_t)(b * M_ + m0) * DOUT_;
    #pragma unroll
    for (int d0 = 0; d0 < DOUT_; d0 += 16) {
        // B[v = 16*hi+e, d = d0+nn] = weff[d*32 + v]: 16 contiguous f16 per lane
        const v16h wb = *(const v16h*)(&weff[(d0 + nn) * DV_ + 16 * hi]);
        const float bias = bo[d0 + nn];
        v8f c;
        #pragma unroll
        for (int g = 0; g < 8; ++g) c[g] = bias;   // bias folded into accumulator
        c = __builtin_amdgcn_wmma_f32_16x16x32_f16(false, ha, false, wb, (short)0, c, false, false);
        #pragma unroll
        for (int g = 0; g < 8; ++g)
            orow[(size_t)(g + 8 * hi) * DOUT_ + d0 + nn] = c[g];
    }
}

extern "C" void kernel_launch(void* const* d_in, const int* in_sizes, int n_in,
                              void* d_out, int out_size, void* d_ws, size_t ws_size,
                              hipStream_t stream) {
    const float* q  = (const float*)d_in[0];
    const float* k  = (const float*)d_in[1];
    const float* v  = (const float*)d_in[2];
    const float* qp = (const float*)d_in[3];
    const float* kp = (const float*)d_in[4];
    const float* wo = (const float*)d_in[5];
    const float* bo = (const float*)d_in[6];
    float* out = (float*)d_out;

    const size_t nK = (size_t)B_ * N_ * DK_;          // 4,194,304 elems
    const size_t nV = (size_t)B_ * N_ * DV_;          //   524,288 elems
    const size_t need = (nK + nV) * sizeof(_Float16); // ~9 MB

    dim3 grid(B_ * (M_ / 16) / 8);   // 128 blocks
    dim3 block(256);                 // 8 waves, wave32

    if (ws_size >= need) {
        _Float16* kf16  = (_Float16*)d_ws;
        _Float16* vf16t = kf16 + nK;
        convert_k_kernel<<<(int)(nK / 8 / 256), 256, 0, stream>>>(k, kf16);
        convert_v_kernel<<<(int)(nV / 256), 256, 0, stream>>>(v, vf16t);
        mha_flash_kernel<true><<<grid, block, 0, stream>>>(q, k, v, qp, kp, wo, bo,
                                                           kf16, vf16t, out);
    } else {
        mha_flash_kernel<false><<<grid, block, 0, stream>>>(q, k, v, qp, kp, wo, bo,
                                                            nullptr, nullptr, out);
    }
}
